// CliffordSharedSimplicialMPNN_GWL_2774548873306
// MI455X (gfx1250) — matrix operand order repaired and checked by reading, not compiled, not run on hardware
//
#include <hip/hip_runtime.h>

// ---------------------------------------------------------------------------
// Clifford simplicial MPNN for MI455X (gfx1250, wave32, WMMA).
// Every cemlp block = 3 per-blade GEMMs [B x FIN] * [FIN x FOUT] (4 distinct
// grade weight matrices) + cheap elementwise glue, fused into one kernel:
// 8 waves per workgroup, wave w owns blade w, WMMA f32<-bf16 16x16x32 tiles
// (K pads 28->32, 56->64).
//   * B (weight) fragments are built ONCE per wave, straight from global
//     memory, and live in VGPRs for the whole grid-stride tile loop.
//   * A (data) operands are staged in LDS in bf16 already in the WMMA lane
//     layout, so a fragment load is two ds_load_b128.
//   * Biases ride in the K-padding: A channel k==FIN carries 1.0 on blade 0,
//     the matching B row carries the bias vector -> the matrix pipe adds the
//     bias, and every WMMA epilogue store is unconditional (no exec masks).
//   * The post-silu A fragment is loaded once and reused by the right- and
//     left- mvlinear WMMAs.
// Input modes fuse the edge gather (x[dst]-x[src]) / node concat([x,agg]);
// output modes fuse segment_sum scatter (atomicAdd) and the residual add.
// ---------------------------------------------------------------------------

typedef __attribute__((ext_vector_type(16))) __bf16 v16bf;
typedef __attribute__((ext_vector_type(8)))  float  v8f;

#define CEPS 1e-6f

// Blade grades for order [1, e1, e2, e3, e12, e13, e23, e123]
__constant__ int c_G[8] = {0, 1, 1, 1, 2, 2, 2, 3};
// Cayley table (Cl(3,0)): blade i * blade j -> blade c_CK[i*8+j], sign c_CS.
__constant__ int c_CK[64] = {
    0,1,2,3,4,5,6,7,
    1,0,4,5,2,3,7,6,
    2,4,0,6,1,7,3,5,
    3,5,6,0,7,1,2,4,
    4,2,1,7,0,6,5,3,
    5,3,7,1,6,0,4,2,
    6,7,3,2,5,4,0,1,
    7,6,5,4,3,2,1,0};
__constant__ float c_CS[64] = {
    1, 1, 1, 1, 1, 1, 1, 1,
    1, 1, 1, 1, 1, 1, 1, 1,
    1,-1, 1, 1,-1,-1, 1,-1,
    1,-1,-1, 1, 1,-1,-1, 1,
    1,-1, 1, 1,-1,-1, 1,-1,
    1,-1,-1, 1, 1,-1,-1, 1,
    1, 1,-1, 1,-1, 1,-1,-1,
    1, 1,-1, 1,-1, 1,-1,-1};

struct BlockParams {
  const float* lin_b;   const float* lin_w;   const float* ln_a;
  const float* left_b;  const float* left_w;  const float* norm_a;
  const float* right_w; const float* sgp_w;   const float* silu_a;
  const float* silu_b;
};

enum { IM_PLAIN = 0, IM_GDIFF = 1, IM_CONCAT = 2 };
enum { OM_STORE = 0, OM_SCATTER = 1, OM_RESID = 2 };

__host__ __device__ constexpr int kpl_of(int fin) { return ((fin + 31) / 32) * 32; }
__host__ __device__ constexpr size_t smem_of(int fin) {
  int KB = kpl_of(fin) / 32;
  size_t bf = (size_t)(KB + 1) * 4096;                  // Xa + Ha (bf16)
  size_t f  = 3 * 4096 + 512 + 16 + 2048 + 384 + 32;    // H,XR,P,NC,NI,sW,scalars
  return bf * 2 + f * 4;
}

__device__ inline float sigm(float x) { return 1.f / (1.f + __expf(-x)); }

// Build a 32x16 (K x N) bf16 B fragment for grade g straight from the global
// weight tensor w[n][k][4 grades].  Zero padded outside [fin x fout]; row
// K==fin optionally carries the bias vector (matched by a 1.0 in the A data).
// Layout: lanes 0-15 hold K=0..15 (half h = K), lanes 16-31 hold K=16..31.
__device__ inline v16bf build_B(const float* __restrict__ w,
                                const float* __restrict__ bias, int fin,
                                int fout, int g, int kofs, int nt, int lane) {
  v16bf b;
  const int N = (lane & 15) + (nt << 4);
  const int kb = kofs + ((lane >> 4) << 4);
#pragma unroll
  for (int h = 0; h < 16; ++h) {
    const int K = kb + h;
    float v = 0.f;
    if (N < fout) {
      if (K < fin)                      v = w[(N * fin + K) * 4 + g];
      else if (K == fin && bias != nullptr) v = bias[N];
    }
    b[h] = (__bf16)v;
  }
  return b;
}

// A-fragment addressing: element (M item, K channel) lives at
// lane = M + 16*((K>>4)&1), half = K&15 within one 32-K block.
__device__ inline int a_lane(int M, int kk) { return M + (((kk >> 4) & 1) << 4); }
__device__ inline int a_half(int kk) { return kk & 15; }

// ---------------------------------------------------------------------------
// Fused cemlp block: y = mvlayernorm(ln_a, sgp(mvsilu(silu, mvlinear(lin,x))))
// ---------------------------------------------------------------------------
template <int FIN>
__global__ __launch_bounds__(256) void cemlp_block_kernel(
    const float* __restrict__ inA, const float* __restrict__ inB,
    const int* __restrict__ sIdx, const int* __restrict__ dIdx,
    float* __restrict__ out, const float* __restrict__ resBase,
    BlockParams bp, int B, int FOUT, int strideA, int strideB,
    int inMode, int outMode) {
  constexpr int KPL = kpl_of(FIN);
  constexpr int KB  = KPL / 32;
  static_assert(FIN < KPL, "need a spare K slot for the bias channel");

  const int tid  = threadIdx.x;
  const int lane = tid & 31;
  const int wave = tid >> 5;  // wave == blade index (8 waves)

  extern __shared__ char smem[];
  __bf16* Xa = (__bf16*)smem;          // [KB][8 blades][32 lanes][16] A staging
  __bf16* Ha = Xa + KB * 4096;         // [8][32][16]  post-silu A staging
  float*  H  = (float*)(Ha + 4096);    // [16][32][8]  post-lin / post-silu f32
  float*  XR = H + 4096;               // [16][32][8]
  float*  P  = XR + 4096;              // [16][32][8]
  float*  NC = P + 4096;               // [16][32]
  float*  NI = NC + 512;               // [16]
  float*  sW = NI + 16;                // [32][64] sgp weights (zero padded)
  float*  sSa = sW + 2048;             // [32][4]
  float*  sSb = sSa + 128;
  float*  sNa = sSb + 128;
  float*  sLn = sNa + 128;             // [32]

  // ---- stage scalar params (zero padded to 32 channels) ----
  for (int i = tid; i < 2048; i += 256) sW[i] = (i < FOUT * 64) ? bp.sgp_w[i] : 0.f;
  for (int i = tid; i < 128; i += 256) {
    bool ok = i < FOUT * 4;
    sSa[i] = ok ? bp.silu_a[i] : 0.f;
    sSb[i] = ok ? bp.silu_b[i] : 0.f;
    sNa[i] = ok ? bp.norm_a[i] : 0.f;
  }
  if (tid < 32) sLn[tid] = (tid < FOUT) ? bp.ln_a[tid] : 0.f;

  // ---- per-wave weight fragments, resident in VGPRs for the whole loop ----
  const int grade = c_G[wave];
  v16bf linB[2][KB], rightB[2], leftB[2];
#pragma unroll
  for (int nt = 0; nt < 2; ++nt) {
#pragma unroll
    for (int kb = 0; kb < KB; ++kb)
      linB[nt][kb] =
          build_B(bp.lin_w, bp.lin_b, FIN, FOUT, grade, kb * 32, nt, lane);
    rightB[nt] = build_B(bp.right_w, nullptr,   FOUT, FOUT, grade, 0, nt, lane);
    leftB[nt]  = build_B(bp.left_w,  bp.left_b, FOUT, FOUT, grade, 0, nt, lane);
  }
  __syncthreads();

  const int ntl = (B + 15) >> 4;
  for (int tile = blockIdx.x; tile < ntl; tile += gridDim.x) {
    const int i0 = tile << 4;

    // prefetch next tile's input rows (global_prefetch_b8 hint)
    if (tid < 16 && inMode != IM_CONCAT) {
      int gi = ((tile + gridDim.x) << 4) + tid;
      if (gi < B) {
        if (inMode == IM_PLAIN) {
          __builtin_prefetch(&inA[(size_t)gi * strideA], 0, 1);
        } else {
          __builtin_prefetch(&inA[(size_t)sIdx[gi] * strideA], 0, 1);
          __builtin_prefetch(&inA[(size_t)dIdx[gi] * strideA], 0, 1);
        }
      }
    }

    // ---- phase 0: stage input tile as bf16 in WMMA A layout (float4 IO) ----
    // unit = (item, channel, 4-blade group); bias channel k==FIN gets 1.0 on
    // blade 0 so the lin WMMA adds lin_b itself.
    for (int i = tid; i < 16 * KPL * 2; i += 256) {
      int it = i / (KPL * 2), r = i - it * (KPL * 2), k = r >> 1, bg = r & 1;
      int gi = i0 + it;
      float4 v = make_float4(0.f, 0.f, 0.f, 0.f);
      if (gi < B && k < FIN) {
        if (inMode == IM_PLAIN) {
          v = *(const float4*)&inA[(size_t)gi * strideA + k * 8 + bg * 4];
        } else if (inMode == IM_GDIFF) {
          int s = sIdx[gi], d = dIdx[gi];
          float4 vd = *(const float4*)&inA[(size_t)d * strideA + k * 8 + bg * 4];
          float4 vs = *(const float4*)&inA[(size_t)s * strideA + k * 8 + bg * 4];
          v = make_float4(vd.x - vs.x, vd.y - vs.y, vd.z - vs.z, vd.w - vs.w);
        } else {  // concat [x, agg]
          v = (k < FIN / 2)
                  ? *(const float4*)&inA[(size_t)gi * strideA + k * 8 + bg * 4]
                  : *(const float4*)&inB[(size_t)gi * strideB +
                                         (k - FIN / 2) * 8 + bg * 4];
        }
      }
      if (k == FIN && bg == 0 && gi < B) v.x = 1.f;  // bias channel, blade 0
      int kb = k >> 5, kk = k & 31;
      int base = ((kb * 8 + bg * 4) * 32 + a_lane(it, kk)) * 16 + a_half(kk);
      Xa[base]             = (__bf16)v.x;
      Xa[base + 32 * 16]   = (__bf16)v.y;
      Xa[base + 2 * 32 * 16] = (__bf16)v.z;
      Xa[base + 3 * 32 * 16] = (__bf16)v.w;
    }
    __syncthreads();

    // ---- phase 1: lin mvlinear via WMMA (bias folded in); branchless store --
    {
      v16bf a[KB];
#pragma unroll
      for (int kb = 0; kb < KB; ++kb)
        a[kb] = *(const v16bf*)&Xa[((kb * 8 + wave) * 32 + lane) * 16];
#pragma unroll
      for (int nt = 0; nt < 2; ++nt) {
        v8f acc = {0.f, 0.f, 0.f, 0.f, 0.f, 0.f, 0.f, 0.f};
#pragma unroll
        for (int kb = 0; kb < KB; ++kb)
          acc = __builtin_amdgcn_wmma_f32_16x16x32_bf16(
              false, a[kb], false, linB[nt][kb], (short)0, acc, false, false);
        const int it0 = (lane >> 4) << 3;
        const int ch  = (lane & 15) + (nt << 4);
#pragma unroll
        for (int r = 0; r < 8; ++r)
          H[((it0 + r) * 32 + ch) * 8 + wave] = acc[r];  // pad cols exactly 0
      }
    }
    __syncthreads();

    // ---- phase 2: mvsilu (in place on H) + bf16 A staging for sgp ----
    for (int e = tid; e < 512; e += 256) {
      int it = e >> 5, ch = e & 31;
      float x[8];
#pragma unroll
      for (int b = 0; b < 8; ++b) x[b] = H[(it * 32 + ch) * 8 + b];
      float inv[4];
      inv[0] = x[0];
      inv[1] = x[1] * x[1] + x[2] * x[2] + x[3] * x[3];
      inv[2] = x[4] * x[4] + x[5] * x[5] + x[6] * x[6];
      inv[3] = x[7] * x[7];
      float gate[4];
#pragma unroll
      for (int g = 0; g < 4; ++g)
        gate[g] = sigm(sSa[ch * 4 + g] * inv[g] + sSb[ch * 4 + g]);
      int la = a_lane(it, ch), hf = a_half(ch);
#pragma unroll
      for (int b = 0; b < 8; ++b) {
        float v = x[b] * gate[c_G[b]];
        H[(it * 32 + ch) * 8 + b] = v;
        Ha[(b * 32 + la) * 16 + hf] = (__bf16)v;
      }
      // left-bias channel: 1.0 on blade 0 at ch==FOUT (right B row there is 0)
      if (ch == FOUT) Ha[la * 16 + hf] = (__bf16)1.f;
    }
    __syncthreads();

    // post-silu A fragment: load once, reuse for right- and left- WMMAs
    v16bf ah = *(const v16bf*)&Ha[(wave * 32 + lane) * 16];

    // ---- phase 3: sgp.right mvlinear via WMMA; branchless store ----
#pragma unroll
    for (int nt = 0; nt < 2; ++nt) {
      v8f acc = {0.f, 0.f, 0.f, 0.f, 0.f, 0.f, 0.f, 0.f};
      acc = __builtin_amdgcn_wmma_f32_16x16x32_bf16(
          false, ah, false, rightB[nt], (short)0, acc, false, false);
      const int it0 = (lane >> 4) << 3;
      const int ch  = (lane & 15) + (nt << 4);
#pragma unroll
      for (int r = 0; r < 8; ++r)
        XR[((it0 + r) * 32 + ch) * 8 + wave] = acc[r];
    }
    __syncthreads();

    // ---- phase 4: grade_norm (in place on XR) ----
    for (int e = tid; e < 512; e += 256) {
      int it = e >> 5, ch = e & 31;
      float x[8];
#pragma unroll
      for (int b = 0; b < 8; ++b) x[b] = XR[(it * 32 + ch) * 8 + b];
      float q[4];
      q[0] = x[0] * x[0];
      q[1] = x[1] * x[1] + x[2] * x[2] + x[3] * x[3];
      q[2] = x[4] * x[4] + x[5] * x[5] + x[6] * x[6];
      q[3] = x[7] * x[7];
      float nrm[4];
#pragma unroll
      for (int g = 0; g < 4; ++g) {
        float n = sqrtf(fmaxf(q[g], 1e-12f));
        nrm[g] = sigm(sNa[ch * 4 + g]) * (n - 1.f) + 1.f;
      }
#pragma unroll
      for (int b = 0; b < 8; ++b)
        XR[(it * 32 + ch) * 8 + b] = x[b] / (nrm[c_G[b]] + CEPS);
    }
    __syncthreads();

    // ---- phase 5: weighted geometric product (sparse Cayley, 64 terms) ----
    for (int e = tid; e < 512; e += 256) {
      int it = e >> 5, ch = e & 31;
      const float* w = sW + ch * 64;  // [g1][g2][g3], zero padded rows
      float S[8], R[8], acc[8];
#pragma unroll
      for (int b = 0; b < 8; ++b) {
        S[b] = H[(it * 32 + ch) * 8 + b];
        R[b] = XR[(it * 32 + ch) * 8 + b];
        acc[b] = 0.f;
      }
#pragma unroll
      for (int i = 0; i < 8; ++i)
#pragma unroll
        for (int j = 0; j < 8; ++j) {
          int k = c_CK[i * 8 + j];
          acc[k] += S[i] * R[j] * c_CS[i * 8 + j] *
                    w[(c_G[i] * 4 + c_G[j]) * 4 + c_G[k]];
        }
#pragma unroll
      for (int b = 0; b < 8; ++b) P[(it * 32 + ch) * 8 + b] = acc[b];
    }
    __syncthreads();

    // ---- phase 6: sgp.left WMMA (bias folded), merge (left+prod)*rsqrt2 ----
#pragma unroll
    for (int nt = 0; nt < 2; ++nt) {
      v8f acc = {0.f, 0.f, 0.f, 0.f, 0.f, 0.f, 0.f, 0.f};
      acc = __builtin_amdgcn_wmma_f32_16x16x32_bf16(
          false, ah, false, leftB[nt], (short)0, acc, false, false);
      const int it0 = (lane >> 4) << 3;
      const int ch  = (lane & 15) + (nt << 4);
#pragma unroll
      for (int r = 0; r < 8; ++r) {
        int o = ((it0 + r) * 32 + ch) * 8 + wave;
        P[o] = (P[o] + acc[r]) * 0.70710678118654752f;
      }
    }
    __syncthreads();

    // ---- phase 7: mvlayernorm + fused output ----
    for (int e = tid; e < 16 * FOUT; e += 256) {
      int it = e / FOUT, ch = e - it * FOUT;
      float s = 0.f;
#pragma unroll
      for (int b = 0; b < 8; ++b) {
        float v = P[(it * 32 + ch) * 8 + b];
        s += v * v;
      }
      NC[it * 32 + ch] = sqrtf(fmaxf(s, 1e-12f));
    }
    __syncthreads();
    if (tid < 16) {
      float s = 0.f;
      for (int ch = 0; ch < FOUT; ++ch) s += NC[tid * 32 + ch];
      NI[tid] = s / (float)FOUT + CEPS;
    }
    __syncthreads();
    if (outMode == OM_STORE) {
      for (int e = tid; e < 16 * FOUT; e += 256) {
        int it = e / FOUT, ch = e - it * FOUT, gi = i0 + it;
        if (gi < B) {
          float scale = sLn[ch] / NI[it];
          size_t o0 = (size_t)gi * FOUT * 8 + ch * 8;
#pragma unroll
          for (int b = 0; b < 8; ++b)
            out[o0 + b] = P[(it * 32 + ch) * 8 + b] * scale;
        }
      }
    } else if (outMode == OM_SCATTER) {
      for (int e = tid; e < 16 * FOUT; e += 256) {
        int it = e / FOUT, ch = e - it * FOUT, gi = i0 + it;
        if (gi < B) {
          float scale = sLn[ch] / NI[it];
          size_t o0 = (size_t)dIdx[gi] * FOUT * 8 + ch * 8;
#pragma unroll
          for (int b = 0; b < 8; ++b)
            atomicAdd(&out[o0 + b], P[(it * 32 + ch) * 8 + b] * scale);
        }
      }
    } else {  // residual
      for (int e = tid; e < 16 * FOUT; e += 256) {
        int it = e / FOUT, ch = e - it * FOUT, gi = i0 + it;
        if (gi < B) {
          float scale = sLn[ch] / NI[it];
          size_t o0 = (size_t)gi * FOUT * 8 + ch * 8;
#pragma unroll
          for (int b = 0; b < 8; ++b)
            out[o0 + b] = resBase[o0 + b] + P[(it * 32 + ch) * 8 + b] * scale;
        }
      }
    }
    __syncthreads();  // protect LDS before next tile
  }
}

// ---------------------------------------------------------------------------
// Small helper kernels
// ---------------------------------------------------------------------------
__global__ void k_zero(float* p, int n) {
  int i = blockIdx.x * blockDim.x + threadIdx.x;
  if (i < n) p[i] = 0.f;
}
__global__ void k_graph_sum(const float* pos, const int* batch, float* gsum,
                            float* gcnt, int N) {
  int i = blockIdx.x * blockDim.x + threadIdx.x;
  if (i < N) {
    int g = batch[i];
    atomicAdd(&gsum[g * 3 + 0], pos[i * 3 + 0]);
    atomicAdd(&gsum[g * 3 + 1], pos[i * 3 + 1]);
    atomicAdd(&gsum[g * 3 + 2], pos[i * 3 + 2]);
    atomicAdd(&gcnt[g], 1.f);
  }
}
__global__ void k_loc(const float* pos, const int* batch, const float* gsum,
                      const float* gcnt, float* loc, int N) {
  int i = blockIdx.x * blockDim.x + threadIdx.x;
  if (i < N) {
    int g = batch[i];
    float c = fmaxf(gcnt[g], 1.f);
    for (int d = 0; d < 3; ++d)
      loc[i * 3 + d] = pos[i * 3 + d] - gsum[g * 3 + d] / c;
  }
}
__global__ void k_embed(const float* loc, const int* x_ind, float* eb, int N) {
  int e = blockIdx.x * blockDim.x + threadIdx.x;  // over N*3
  if (e < N * 3) {
    int idx = x_ind[e];
    float* o = eb + (size_t)e * 8;
    o[0] = 0.f;
    o[1] = loc[idx * 3 + 0];
    o[2] = loc[idx * 3 + 1];
    o[3] = loc[idx * 3 + 2];
    o[4] = 0.f; o[5] = 0.f; o[6] = 0.f; o[7] = 0.f;
  }
}
__global__ void k_emb0(const float* eb, const float* w, const float* b,
                       float* x0, int N) {
  int e = blockIdx.x * blockDim.x + threadIdx.x;  // over N*28*8
  if (e < N * 224) {
    int n = e / 224, r = e - n * 224, c = r >> 3, bl = r & 7;
    float v = w[c] * eb[(size_t)n * 24 + bl];  // j = 0 slot of embed
    if (bl == 0) v += b[c];
    x0[e] = v;
  }
}
__global__ void k_select(const int* nt, const float* x0, const float* x1,
                         const float* x2, float* x, int N) {
  int e = blockIdx.x * blockDim.x + threadIdx.x;
  if (e < N * 224) {
    int t = nt[e / 224];
    x[e] = (t == 0) ? x0[e] : ((t == 1) ? x1[e] : x2[e]);
  }
}
__global__ void k_pool(const float* x, const int* batch, float* pooled, int N) {
  int e = blockIdx.x * blockDim.x + threadIdx.x;
  if (e < N * 224) {
    int n = e / 224, f = e - n * 224;
    atomicAdd(&pooled[(size_t)batch[n] * 224 + f], x[e]);
  }
}
__global__ void k_pool_div(float* pooled, const float* gcnt, int G) {
  int e = blockIdx.x * blockDim.x + threadIdx.x;
  if (e < G * 224) pooled[e] /= fmaxf(gcnt[e / 224], 1.f);
}

// ---------------------------------------------------------------------------
// Host orchestration
// ---------------------------------------------------------------------------
static inline BlockParams bp_at(void* const* d, int base) {
  BlockParams p;  // pytree (sorted-key) leaf order within a cemlp block:
  p.lin_b   = (const float*)d[base + 0];  // lin.b
  p.lin_w   = (const float*)d[base + 1];  // lin.w
  p.ln_a    = (const float*)d[base + 2];  // ln_a
  p.left_b  = (const float*)d[base + 3];  // sgp.left.b
  p.left_w  = (const float*)d[base + 4];  // sgp.left.w
  p.norm_a  = (const float*)d[base + 5];  // sgp.norm_a
  p.right_w = (const float*)d[base + 6];  // sgp.right.w
  p.sgp_w   = (const float*)d[base + 7];  // sgp.w
  p.silu_a  = (const float*)d[base + 8];  // silu.a
  p.silu_b  = (const float*)d[base + 9];  // silu.b
  return p;
}

template <int FIN>
static void run_cemlp(const float* inA, int strideA, const float* inB, int strideB,
                      const int* sIdx, const int* dIdx, float* out,
                      const float* resBase, const BlockParams& bp, int B, int FOUT,
                      int inMode, int outMode, hipStream_t stream) {
  int blocks = (B + 15) / 16;
  if (blocks > 2048) blocks = 2048;
  cemlp_block_kernel<FIN><<<blocks, 256, (unsigned)smem_of(FIN), stream>>>(
      inA, inB, sIdx, dIdx, out, resBase, bp, B, FOUT, strideA, strideB,
      inMode, outMode);
}

extern "C" void kernel_launch(void* const* d_in, const int* in_sizes, int n_in,
                              void* d_out, int out_size, void* d_ws, size_t ws_size,
                              hipStream_t stream) {
  const float* pos        = (const float*)d_in[0];
  const int*   x_ind      = (const int*)d_in[1];
  const int*   node_types = (const int*)d_in[2];
  const int*   edge_index = (const int*)d_in[3];
  const int*   batch      = (const int*)d_in[4];
  const int N  = in_sizes[0] / 3;
  const int E  = in_sizes[3] / 2;
  const int NG = 64;
  const int F8 = 28 * 8;  // 224 floats per multivector feature

  // workspace carve (floats)
  float* w      = (float*)d_ws;
  float* gsum   = w;  w += NG * 3;
  float* gcnt   = w;  w += NG;
  float* loc    = w;  w += (size_t)N * 3;
  float* eb     = w;  w += (size_t)N * 24;         // embed(loc[x_ind]) [N][3][8]
  float* b0     = w;  w += (size_t)N * F8;         // x0, later agg
  float* b1     = w;  w += (size_t)N * F8;         // x1, later node tmp
  float* b2     = w;  w += (size_t)N * F8;         // emb2 intermediate
  float* b3     = w;  w += (size_t)N * F8;         // x2
  float* bx     = w;  w += (size_t)N * F8;         // node features x
  float* ebuf   = w;  w += (size_t)E * F8;         // edge intermediate
  float* pooled = w;  w += NG * F8;

  const int T = 256;
  auto nb = [](long long n) { return (int)((n + 255) / 256); };

  // graph means -> loc
  k_zero<<<1, T, 0, stream>>>(gsum, NG * 4);
  k_graph_sum<<<nb(N), T, 0, stream>>>(pos, batch, gsum, gcnt, N);
  k_loc<<<nb(N), T, 0, stream>>>(pos, batch, gsum, gcnt, loc, N);
  k_embed<<<nb((long long)N * 3), T, 0, stream>>>(loc, x_ind, eb, N);

  // embeddings (param leaves start at d_in[6]; emb0.b=6, emb0.w=7)
  k_emb0<<<nb((long long)N * F8), T, 0, stream>>>(
      eb, (const float*)d_in[7], (const float*)d_in[6], b0, N);
  run_cemlp<2>(eb, 24, nullptr, 0, nullptr, nullptr, b1, nullptr,
               bp_at(d_in, 8), N, 28, IM_PLAIN, OM_STORE, stream);
  run_cemlp<3>(eb, 24, nullptr, 0, nullptr, nullptr, b2, nullptr,
               bp_at(d_in, 18), N, 28, IM_PLAIN, OM_STORE, stream);
  run_cemlp<28>(b2, F8, nullptr, 0, nullptr, nullptr, b3, nullptr,
                bp_at(d_in, 28), N, 28, IM_PLAIN, OM_STORE, stream);
  k_select<<<nb((long long)N * F8), T, 0, stream>>>(node_types, b0, b1, b3, bx, N);

  // message passing (layers start at leaf 38; 40 leaves per layer)
  const int* src = edge_index;
  const int* dst = edge_index + E;
  for (int l = 0; l < 3; ++l) {
    int base = 38 + 40 * l;
    // edge cemlp block 1: gather x[dst]-x[src]
    run_cemlp<28>(bx, F8, nullptr, 0, src, dst, ebuf, nullptr,
                  bp_at(d_in, base + 0), E, 28, IM_GDIFF, OM_STORE, stream);
    // edge cemlp block 2 with fused segment_sum scatter into agg (b0)
    k_zero<<<nb((long long)N * F8), T, 0, stream>>>(b0, N * F8);
    run_cemlp<28>(ebuf, F8, nullptr, 0, nullptr, dst, b0, nullptr,
                  bp_at(d_in, base + 10), E, 28, IM_PLAIN, OM_SCATTER, stream);
    // node cemlp block 1: concat([x, agg])
    run_cemlp<56>(bx, F8, b0, F8, nullptr, nullptr, b1, nullptr,
                  bp_at(d_in, base + 20), N, 28, IM_CONCAT, OM_STORE, stream);
    // node cemlp block 2 with fused residual: x = x + f(...)
    run_cemlp<28>(b1, F8, nullptr, 0, nullptr, nullptr, bx, bx,
                  bp_at(d_in, base + 30), N, 28, IM_PLAIN, OM_RESID, stream);
  }

  // pooling + projection head (proj leaves at 158; FOUT=2 -> (64,2,8))
  k_zero<<<nb(NG * F8), T, 0, stream>>>(pooled, NG * F8);
  k_pool<<<nb((long long)N * F8), T, 0, stream>>>(bx, batch, pooled, N);
  k_pool_div<<<nb(NG * F8), T, 0, stream>>>(pooled, gcnt, NG);
  run_cemlp<28>(pooled, F8, nullptr, 0, nullptr, nullptr, (float*)d_out, nullptr,
                bp_at(d_in, 158), NG, 2, IM_PLAIN, OM_STORE, stream);
}
